// MetaMulti_39444979647205
// MI455X (gfx1250) — matrix operand order repaired and checked by reading, not compiled, not run on hardware
//
#include <hip/hip_runtime.h>
#include <stdint.h>

// ---------------------------------------------------------------------------
// MeshGraphNet-style GNN for MI455X (gfx1250).
// All MLPs run as bf16 WMMA (v_wmma_f32_16x16x32_bf16) tall-skinny GEMMs:
// one wave32 owns a 16-row tile; activations staged per-wave in LDS (bf16);
// weights pre-packed into WMMA B-fragment layout (global, L2-resident).
// Memory-bound by e-stream (~3 GB over 3 layers -> ~130us @ 23.3 TB/s).
// ---------------------------------------------------------------------------

typedef __bf16 bf16;
typedef __attribute__((ext_vector_type(16))) __bf16 v16bf;
typedef __attribute__((ext_vector_type(8)))  __bf16 v8bf;
typedef __attribute__((ext_vector_type(8)))  float  v8f;

#define WAVES 4          // 4 waves per 128-thread block (one WGP-friendly)
#define KMAX  320        // largest MLP input width (glob MLP: 5*64)

__device__ __forceinline__ int laneId() { return threadIdx.x & 31; }

// --- A fragment (16x32 bf16, documented CDNA5 layout) from LDS -------------
// lane l: row m = l&15, half h = l>>4; VGPR0-3 hold K = 8h..8h+7,
// VGPR4-7 hold K = 16+8h..16+8h+7 (two contiguous 16B chunks).
__device__ __forceinline__ v16bf load_a_frag(const bf16* arow, int k0) {
  v8bf lo = *(const v8bf*)(arow + k0);
  v8bf hi = *(const v8bf*)(arow + k0 + 16);
  v16bf a;
#pragma unroll
  for (int i = 0; i < 8; ++i) { a[i] = lo[i]; a[8 + i] = hi[i]; }
  return a;
}

// --- GEMM: C(16xNB*16) += A(16xK) @ W(KxNB*16), bf16 in / f32 acc ----------
// W is pre-packed: block (kb,nb) = 1KB; lane holds 8 dwords (16 bf16).
template <int NB>
__device__ __forceinline__ void gemm_tile(const bf16* __restrict__ A, int strideA,
                                          int K, const uint8_t* __restrict__ W,
                                          v8f acc[NB]) {
  const int lane = laneId();
  const int m = lane & 15, h = lane >> 4;
  const bf16* arow = A + m * strideA + 8 * h;
#pragma unroll
  for (int nb = 0; nb < NB; ++nb) {
    v8f z = {0.f, 0.f, 0.f, 0.f, 0.f, 0.f, 0.f, 0.f};
    acc[nb] = z;
  }
  const int nk = K >> 5;
  for (int kb = 0; kb < nk; ++kb) {
    v16bf a = load_a_frag(arow, kb * 32);
    const v16bf* wb = (const v16bf*)(W + (size_t)kb * NB * 1024);
#pragma unroll
    for (int nb = 0; nb < NB; ++nb) {
      v16bf b = wb[nb * 32 + lane];
      acc[nb] = __builtin_amdgcn_wmma_f32_16x16x32_bf16(
          false, a, false, b, (short)0, acc[nb], false, false);
    }
  }
}

// bias + relu in C layout, store bf16 tile (16x64) to LDS for next layer
__device__ __forceinline__ void act_relu_store(v8f acc[4], const float* __restrict__ bias,
                                               bf16* __restrict__ ldsH) {
  const int lane = laneId();
  const int nlo = lane & 15, h = lane >> 4;
#pragma unroll
  for (int nb = 0; nb < 4; ++nb) {
    float bv = bias[nb * 16 + nlo];
#pragma unroll
    for (int r = 0; r < 8; ++r) {
      float v = fmaxf(acc[nb][r] + bv, 0.f);
      ldsH[(r + 8 * h) * 64 + nb * 16 + nlo] = (bf16)v;
    }
  }
}

// --- full MLP blob: K0->64 relu, 64->64 relu, 64->64 relu, LN, 64->out -----
// Packed blob layout: W0 | W1 | W2 | W3 | b0 b1 b2 g be b3 (floats)
template <int NB3>
__device__ __forceinline__ void run_mlp(const uint8_t* __restrict__ blob, int K0,
                                        const bf16* __restrict__ ldsIn, int strideIn,
                                        bf16* __restrict__ ldsH, v8f out[4]) {
  const int lane = laneId();
  const int nlo = lane & 15, h = lane >> 4;
  const uint8_t* w0 = blob;
  const uint8_t* w1 = blob + (size_t)(K0 >> 5) * 4096;
  const uint8_t* w2 = w1 + 8192;
  const uint8_t* w3 = w2 + 8192;
  const float* fb = (const float*)(w3 + NB3 * 2048);
  const float* b0 = fb;
  const float* b1 = fb + 64;
  const float* b2 = fb + 128;
  const float* g  = fb + 192;
  const float* be = fb + 256;
  const float* b3 = fb + 320;

  v8f acc[4];
  gemm_tile<4>(ldsIn, strideIn, K0, w0, acc);
  act_relu_store(acc, b0, ldsH);
  gemm_tile<4>(ldsH, 64, 64, w1, acc);
  act_relu_store(acc, b1, ldsH);
  gemm_tile<4>(ldsH, 64, 64, w2, acc);

  // bias + relu, then LayerNorm in C layout (rows span 16 lanes of a half)
  float vreg[4][8];
#pragma unroll
  for (int nb = 0; nb < 4; ++nb) {
    float bv = b2[nb * 16 + nlo];
#pragma unroll
    for (int r = 0; r < 8; ++r) vreg[nb][r] = fmaxf(acc[nb][r] + bv, 0.f);
  }
  float sm[8], s2[8];
#pragma unroll
  for (int r = 0; r < 8; ++r) {
    float t = 0.f, t2 = 0.f;
#pragma unroll
    for (int nb = 0; nb < 4; ++nb) { float v = vreg[nb][r]; t += v; t2 += v * v; }
    sm[r] = t; s2[r] = t2;
  }
#pragma unroll
  for (int mask = 1; mask < 16; mask <<= 1) {
#pragma unroll
    for (int r = 0; r < 8; ++r) {
      sm[r] += __shfl_xor(sm[r], mask, 32);
      s2[r] += __shfl_xor(s2[r], mask, 32);
    }
  }
#pragma unroll
  for (int nb = 0; nb < 4; ++nb) {
    float gv = g[nb * 16 + nlo], bev = be[nb * 16 + nlo];
#pragma unroll
    for (int r = 0; r < 8; ++r) {
      float mean = sm[r] * (1.f / 64.f);
      float var  = s2[r] * (1.f / 64.f) - mean * mean;
      float inv  = rsqrtf(var + 1e-5f);
      float v = (vreg[nb][r] - mean) * inv * gv + bev;
      ldsH[(r + 8 * h) * 64 + nb * 16 + nlo] = (bf16)v;
    }
  }
  gemm_tile<NB3>(ldsH, 64, 64, w3, out);
#pragma unroll
  for (int nb = 0; nb < NB3; ++nb) {
    float bv = b3[nb * 16 + nlo];
#pragma unroll
    for (int r = 0; r < 8; ++r) out[nb][r] += bv;
  }
}

// --- float atomics min/max via int-bit ordering trick ----------------------
__device__ __forceinline__ void atomicMaxF(float* a, float v) {
  if (v >= 0.f) atomicMax((int*)a, __float_as_int(v));
  else          atomicMin((unsigned int*)a, __float_as_uint(v));
}
__device__ __forceinline__ void atomicMinF(float* a, float v) {
  if (v >= 0.f) atomicMin((int*)a, __float_as_int(v));
  else          atomicMax((unsigned int*)a, __float_as_uint(v));
}

// ===========================================================================
// Weight packing: f32 row-major (K x Nout) -> bf16 WMMA B-fragment blocks.
// Block (kb,nb) is 32x16; lane l holds col nb*16+(l&15), K = kb*32+(l>>4)*8
// + 2j,2j+1 (+8 for j>=4) in dword j.  Biases/LN params copied as f32.
// ===========================================================================
__global__ void k_pack(const float* __restrict__ W0, const float* __restrict__ W1,
                       const float* __restrict__ W2, const float* __restrict__ W3,
                       const float* __restrict__ b0, const float* __restrict__ b1,
                       const float* __restrict__ b2, const float* __restrict__ b3,
                       const float* __restrict__ g,  const float* __restrict__ be,
                       int kin, int k0, int nout, int nb3, uint8_t* __restrict__ blob) {
  const float* Ws[4] = {W0, W1, W2, W3};
  const int kinA[4] = {kin, 64, 64, 64};
  const int kpad[4] = {k0, 64, 64, 64};
  const int noA[4]  = {64, 64, 64, nout};
  const int nbl[4]  = {4, 4, 4, nb3};
  const int tid = blockIdx.x * blockDim.x + threadIdx.x;
  const int nt  = gridDim.x * blockDim.x;
  size_t off = 0;
  for (int L = 0; L < 4; ++L) {
    const int nk = kpad[L] >> 5, NB = nbl[L];
    const int slots = nk * NB * 32;
    uint32_t* dst = (uint32_t*)(blob + off);
    for (int slot = tid; slot < slots; slot += nt) {
      int lane = slot & 31, blk = slot >> 5;
      int nb = blk % NB, kb = blk / NB;
      int n = nb * 16 + (lane & 15);
      int kbase = kb * 32 + (lane >> 4) * 8;
      const float* W = Ws[L];
      const int KA = kinA[L], NO = noA[L];
#pragma unroll
      for (int j = 0; j < 8; ++j) {
        int ke0 = kbase + 2 * j + (j >= 4 ? 8 : 0), ke1 = ke0 + 1;
        float f0 = (ke0 < KA && n < NO) ? W[ke0 * NO + n] : 0.f;
        float f1 = (ke1 < KA && n < NO) ? W[ke1 * NO + n] : 0.f;
        union { bf16 hh[2]; uint32_t u; } pk;
        pk.hh[0] = (bf16)f0; pk.hh[1] = (bf16)f1;
        dst[blk * 256 + lane * 8 + j] = pk.u;
      }
    }
    off += (size_t)nk * NB * 1024;
  }
  float* fbo = (float*)(blob + off);
  const float* srcs[5] = {b0, b1, b2, g, be};
  const int nfl = 320 + 16 * nb3;
  for (int t = tid; t < nfl; t += nt) {
    float v;
    if (t < 320) v = srcs[t >> 6][t & 63];
    else { int n = t - 320; v = (n < nout) ? b3[n] : 0.f; }
    fbo[t] = v;
  }
}

// --- helpers ---------------------------------------------------------------
__global__ void k_fill(float* __restrict__ p, size_t n, float v) {
  size_t i = (size_t)blockIdx.x * blockDim.x + threadIdx.x;
  size_t st = (size_t)gridDim.x * blockDim.x;
  for (; i < n; i += st) p[i] = v;
}

__global__ void k_count(const int* __restrict__ batch, int N, float* __restrict__ cnt) {
  int i = blockIdx.x * blockDim.x + threadIdx.x;
  if (i < N) atomicAdd(&cnt[batch[i]], 1.f);
}

__global__ void k_xg(const float* __restrict__ x, const int* __restrict__ batch,
                     int N, float* __restrict__ xg) {
  size_t i = (size_t)blockIdx.x * blockDim.x + threadIdx.x;
  size_t total = (size_t)N * 64;
  size_t st = (size_t)gridDim.x * blockDim.x;
  for (; i < total; i += st) {
    int node = (int)(i >> 6), c = (int)(i & 63);
    atomicAdd(&xg[batch[node] * 64 + c], x[i]);
  }
}

// ===========================================================================
// Node encoder: x = MLP(graph_x), K0 = 6 padded to 32
// ===========================================================================
__global__ __launch_bounds__(128) void k_node_enc(const float* __restrict__ gx, int N,
                                                  const uint8_t* __restrict__ blob,
                                                  float* __restrict__ x) {
  __shared__ __attribute__((aligned(16))) bf16 sIn[WAVES][16 * KMAX];
  __shared__ __attribute__((aligned(16))) bf16 sH[WAVES][16 * 64];
  const int wave = threadIdx.x >> 5, lane = laneId();
  const int base = (blockIdx.x * WAVES + wave) * 16;
  if (base >= N) return;
  bf16* A = sIn[wave];
  for (int idx = lane; idx < 16 * 32; idx += 32) {
    int r = idx >> 5, k = idx & 31;
    int row = base + r; if (row >= N) row = N - 1;
    A[idx] = (bf16)((k < 6) ? gx[row * 6 + k] : 0.f);
  }
  v8f out[4];
  run_mlp<4>(blob, 32, A, 32, sH[wave], out);
  const int nlo = lane & 15, h = lane >> 4;
#pragma unroll
  for (int nb = 0; nb < 4; ++nb)
#pragma unroll
    for (int r = 0; r < 8; ++r) {
      int row = base + r + 8 * h;
      if (row < N) x[(size_t)row * 64 + nb * 16 + nlo] = out[nb][r];
    }
}

// ===========================================================================
// Edge encoder: e = MLP([edge_attr, gx[r][0]-gx[c][0], gx[r][3]-gx[c][3]])
// ===========================================================================
__global__ __launch_bounds__(128) void k_edge_enc(const float* __restrict__ gx,
                                                  const int* __restrict__ ei,
                                                  const float* __restrict__ ea,
                                                  int N, int E,
                                                  const uint8_t* __restrict__ blob,
                                                  float* __restrict__ e) {
  __shared__ __attribute__((aligned(16))) bf16 sIn[WAVES][16 * KMAX];
  __shared__ __attribute__((aligned(16))) bf16 sH[WAVES][16 * 64];
  const int wave = threadIdx.x >> 5, lane = laneId();
  const int base = (blockIdx.x * WAVES + wave) * 16;
  if (base >= E) return;
  bf16* A = sIn[wave];
  for (int idx = lane; idx < 16 * 32; idx += 32) A[idx] = (bf16)0.f;
  if (lane < 16) {
    int eg = base + lane; if (eg >= E) eg = E - 1;
    int r = ei[eg], c = ei[E + eg];
    A[lane * 32 + 0] = (bf16)ea[eg];
    A[lane * 32 + 1] = (bf16)(gx[r * 6 + 0] - gx[c * 6 + 0]);
    A[lane * 32 + 2] = (bf16)(gx[r * 6 + 3] - gx[c * 6 + 3]);
  }
  v8f out[4];
  run_mlp<4>(blob, 32, A, 32, sH[wave], out);
  const int nlo = lane & 15, h = lane >> 4;
#pragma unroll
  for (int nb = 0; nb < 4; ++nb)
#pragma unroll
    for (int r = 0; r < 8; ++r) {
      int eg = base + r + 8 * h;
      if (eg < E) e[(size_t)eg * 64 + nb * 16 + nlo] = out[nb][r];
    }
}

// ===========================================================================
// Fused per-layer edge stage: e += edgeMLP([x[row],x[col],e,u[b[row]]]);
// m = node1MLP([x[row], e_new]); agg[col] += m (atomic, L2).
// ===========================================================================
__global__ __launch_bounds__(128) void k_edge_layer(const float* __restrict__ x,
                                                    float* __restrict__ e,
                                                    const float* __restrict__ u,
                                                    const int* __restrict__ ei,
                                                    const int* __restrict__ batch,
                                                    int N, int E,
                                                    const uint8_t* __restrict__ blobE,
                                                    const uint8_t* __restrict__ blobN1,
                                                    float* __restrict__ agg) {
  __shared__ __attribute__((aligned(16))) bf16 sIn[WAVES][16 * KMAX];
  __shared__ __attribute__((aligned(16))) bf16 sH[WAVES][16 * 64];
  __shared__ int sIdx[WAVES][32];
  const int wave = threadIdx.x >> 5, lane = laneId();
  const int base = (blockIdx.x * WAVES + wave) * 16;
  if (base >= E) return;
  if (lane < 16) {
    int eg = base + lane; if (eg >= E) eg = E - 1;
    sIdx[wave][lane]      = ei[eg];
    sIdx[wave][16 + lane] = ei[E + eg];
  }
  bf16* A = sIn[wave];
  for (int r = 0; r < 16; ++r) {
    int row = sIdx[wave][r];
    int cix = sIdx[wave][16 + r];
    int eg = base + r; if (eg >= E) eg = E - 1;
    int gb = batch[row];
    for (int t = lane; t < 64; t += 32) {
      A[r * 256 + t]       = (bf16)x[(size_t)row * 64 + t];
      A[r * 256 + 64 + t]  = (bf16)x[(size_t)cix * 64 + t];
      A[r * 256 + 128 + t] = (bf16)e[(size_t)eg * 64 + t];
      A[r * 256 + 192 + t] = (bf16)u[gb * 64 + t];
    }
  }
  v8f de[4];
  run_mlp<4>(blobE, 256, A, 256, sH[wave], de);
  const int nlo = lane & 15, h = lane >> 4;
#pragma unroll
  for (int nb = 0; nb < 4; ++nb)
#pragma unroll
    for (int r = 0; r < 8; ++r) {
      int eg = base + r + 8 * h;
      if (eg < E) {
        size_t off = (size_t)eg * 64 + nb * 16 + nlo;
        float env = e[off] + de[nb][r];
        e[off] = env;  // each edge owned by exactly one wave -> race-free
        A[(r + 8 * h) * 256 + 64 + nb * 16 + nlo] = (bf16)env;  // node1 input
      }
    }
  v8f m[4];
  run_mlp<4>(blobN1, 128, A, 256, sH[wave], m);  // cols 0..127 = [x[row], e_new]
#pragma unroll
  for (int r = 0; r < 8; ++r) {
    int eg = base + r + 8 * h;
    if (eg < E) {
      int cix = sIdx[wave][16 + r + 8 * h];
#pragma unroll
      for (int nb = 0; nb < 4; ++nb)
        atomicAdd(&agg[(size_t)cix * 64 + nb * 16 + nlo], m[nb][r]);
    }
  }
}

// ===========================================================================
// Node update: x += MLP([x, agg, u[batch]]); accumulate per-graph stats
// ===========================================================================
__global__ __launch_bounds__(128) void k_node2(const float* __restrict__ agg,
                                               const float* __restrict__ u,
                                               const int* __restrict__ batch, int N,
                                               const uint8_t* __restrict__ blob,
                                               float* __restrict__ x,
                                               float* __restrict__ s,
                                               float* __restrict__ sq,
                                               float* __restrict__ mi,
                                               float* __restrict__ ma) {
  __shared__ __attribute__((aligned(16))) bf16 sIn[WAVES][16 * KMAX];
  __shared__ __attribute__((aligned(16))) bf16 sH[WAVES][16 * 64];
  __shared__ int sIdx[WAVES][32];
  const int wave = threadIdx.x >> 5, lane = laneId();
  const int base = (blockIdx.x * WAVES + wave) * 16;
  if (base >= N) return;
  if (lane < 16) {
    int row = base + lane; if (row >= N) row = N - 1;
    sIdx[wave][lane] = batch[row];
  }
  bf16* A = sIn[wave];
  for (int r = 0; r < 16; ++r) {
    int row = base + r; if (row >= N) row = N - 1;
    int gb = sIdx[wave][r];
    for (int t = lane; t < 64; t += 32) {
      A[r * 192 + t]       = (bf16)x[(size_t)row * 64 + t];
      A[r * 192 + 64 + t]  = (bf16)agg[(size_t)row * 64 + t];
      A[r * 192 + 128 + t] = (bf16)u[gb * 64 + t];
    }
  }
  v8f out[4];
  run_mlp<4>(blob, 192, A, 192, sH[wave], out);
  const int nlo = lane & 15, h = lane >> 4;
#pragma unroll
  for (int nb = 0; nb < 4; ++nb)
#pragma unroll
    for (int r = 0; r < 8; ++r) {
      int row = base + r + 8 * h;
      if (row < N) {
        size_t off = (size_t)row * 64 + nb * 16 + nlo;
        float xv = x[off] + out[nb][r];
        x[off] = xv;
        int gb = sIdx[wave][r + 8 * h];
        size_t go = (size_t)gb * 64 + nb * 16 + nlo;
        atomicAdd(&s[go], xv);
        atomicAdd(&sq[go], xv * xv);
        atomicMaxF(&ma[go], xv);
        atomicMinF(&mi[go], xv);
      }
    }
}

// ===========================================================================
// Global update: u += MLP([u, s, mi, ma, sq/cnt - (s/cnt)^2]), K0 = 320
// ===========================================================================
__global__ __launch_bounds__(128) void k_glob(const float* __restrict__ s,
                                              const float* __restrict__ sq,
                                              const float* __restrict__ mi,
                                              const float* __restrict__ ma,
                                              const float* __restrict__ cnt, int G,
                                              const uint8_t* __restrict__ blob,
                                              float* __restrict__ u) {
  __shared__ __attribute__((aligned(16))) bf16 sIn[WAVES][16 * KMAX];
  __shared__ __attribute__((aligned(16))) bf16 sH[WAVES][16 * 64];
  const int wave = threadIdx.x >> 5, lane = laneId();
  const int base = (blockIdx.x * WAVES + wave) * 16;
  if (base >= G) return;
  bf16* A = sIn[wave];
  for (int r = 0; r < 16; ++r) {
    int g = base + r; if (g >= G) g = G - 1;
    float c = fmaxf(cnt[g], 1.f);
    for (int t = lane; t < 64; t += 32) {
      size_t go = (size_t)g * 64 + t;
      float sv = s[go];
      float me = sv / c;
      float sd = sq[go] / c - me * me;
      A[r * 320 + t]       = (bf16)u[go];
      A[r * 320 + 64 + t]  = (bf16)sv;
      A[r * 320 + 128 + t] = (bf16)mi[go];
      A[r * 320 + 192 + t] = (bf16)ma[go];
      A[r * 320 + 256 + t] = (bf16)sd;
    }
  }
  v8f out[4];
  run_mlp<4>(blob, 320, A, 320, sH[wave], out);
  const int nlo = lane & 15, h = lane >> 4;
#pragma unroll
  for (int nb = 0; nb < 4; ++nb)
#pragma unroll
    for (int r = 0; r < 8; ++r) {
      int g = base + r + 8 * h;
      if (g < G) u[(size_t)g * 64 + nb * 16 + nlo] += out[nb][r];
    }
}

// ===========================================================================
// Decoder: out = MLP(LayerNorm(xg; norm_g, norm_b)), Nout = 2 (padded 16)
// ===========================================================================
__global__ __launch_bounds__(128) void k_decoder(const float* __restrict__ xg,
                                                 const float* __restrict__ ng,
                                                 const float* __restrict__ nbv,
                                                 int G, const uint8_t* __restrict__ blob,
                                                 float* __restrict__ out) {
  __shared__ __attribute__((aligned(16))) bf16 sIn[WAVES][16 * KMAX];
  __shared__ __attribute__((aligned(16))) bf16 sH[WAVES][16 * 64];
  __shared__ float sStat[WAVES][32];
  const int wave = threadIdx.x >> 5, lane = laneId();
  const int base = (blockIdx.x * WAVES + wave) * 16;
  if (base >= G) return;
  if (lane < 16) {
    int g = base + lane; if (g >= G) g = G - 1;
    float sm = 0.f, s2 = 0.f;
    for (int c = 0; c < 64; ++c) { float v = xg[g * 64 + c]; sm += v; s2 += v * v; }
    float mean = sm * (1.f / 64.f);
    float var = s2 * (1.f / 64.f) - mean * mean;
    sStat[wave][lane] = mean;
    sStat[wave][16 + lane] = rsqrtf(var + 1e-5f);
  }
  bf16* A = sIn[wave];
  for (int idx = lane; idx < 16 * 64; idx += 32) {
    int r = idx >> 6, c = idx & 63;
    int g = base + r; if (g >= G) g = G - 1;
    float v = (xg[g * 64 + c] - sStat[wave][r]) * sStat[wave][16 + r] * ng[c] + nbv[c];
    A[r * 64 + c] = (bf16)v;
  }
  v8f o[4];
  run_mlp<1>(blob, 64, A, 64, sH[wave], o);
  const int nlo = lane & 15, h = lane >> 4;
#pragma unroll
  for (int r = 0; r < 8; ++r) {
    int g = base + r + 8 * h;
    if (g < G && nlo < 2) out[g * 2 + nlo] = o[0][r];
  }
}

// ===========================================================================
// Host orchestration
// ===========================================================================
extern "C" void kernel_launch(void* const* d_in, const int* in_sizes, int n_in,
                              void* d_out, int out_size, void* d_ws, size_t ws_size,
                              hipStream_t stream) {
  (void)n_in; (void)out_size; (void)ws_size;
  const float* gx    = (const float*)d_in[0];
  const int*   ei    = (const int*)d_in[1];
  const float* ea    = (const float*)d_in[2];
  const int*   batch = (const int*)d_in[3];
  const int N = in_sizes[0] / 6;
  const int E = in_sizes[2];
  const int G = 256;

  // jax pytree flatten order (sorted dict keys); per-MLP: [W0..W3, be, b0..b3, g]
  const int DEC = 4, EENC = 14, NENC = 24, NORMB = 34, NORMG = 35, OPS = 36;
  auto opIdx = [&](int layer, int which) { return OPS + layer * 40 + which * 10; };
  // which: 0=edge, 1=glob, 2=node1, 3=node2 (sorted keys)

  uint8_t* ws = (uint8_t*)d_ws;
  size_t off = 0;
  auto carve = [&](size_t bytes) -> size_t {
    size_t o = off; off = (off + bytes + 255) & ~(size_t)255; return o;
  };
  struct MP { int pbase, kin, k0, nout, nb3; size_t o; };
  MP mp[15]; int mc = 0;
  auto addMlp = [&](int pbase, int kin, int k0, int nout, int nb3) {
    size_t bytes = (size_t)(k0 / 32) * 4096 + 8192 + 8192 +
                   (size_t)nb3 * 2048 + (size_t)(320 + 16 * nb3) * 4;
    mp[mc++] = {pbase, kin, k0, nout, nb3, carve(bytes)};
  };
  addMlp(NENC, 6, 32, 64, 4);                 // 0: node encoder
  addMlp(EENC, 3, 32, 64, 4);                 // 1: edge encoder
  for (int L = 0; L < 3; ++L) {
    addMlp(opIdx(L, 0), 256, 256, 64, 4);     // edge
    addMlp(opIdx(L, 2), 128, 128, 64, 4);     // node1
    addMlp(opIdx(L, 3), 192, 192, 64, 4);     // node2
    addMlp(opIdx(L, 1), 320, 320, 64, 4);     // glob
  }
  addMlp(DEC, 64, 64, 2, 1);                  // 14: decoder

  size_t oX   = carve((size_t)N * 64 * 4);
  size_t oE   = carve((size_t)E * 64 * 4);
  size_t oAgg = carve((size_t)N * 64 * 4);
  size_t oU   = carve((size_t)G * 64 * 4);
  size_t oCnt = carve((size_t)G * 4);
  size_t oS   = carve((size_t)G * 64 * 4);
  size_t oSq  = carve((size_t)G * 64 * 4);
  size_t oMi  = carve((size_t)G * 64 * 4);
  size_t oMa  = carve((size_t)G * 64 * 4);
  size_t oXg  = carve((size_t)G * 64 * 4);

  // 1) pack weights into WMMA B-fragment layout (tiny, L2-resident)
  for (int i = 0; i < mc; ++i) {
    const MP& m = mp[i];
    k_pack<<<16, 256, 0, stream>>>(
        (const float*)d_in[m.pbase + 0], (const float*)d_in[m.pbase + 1],
        (const float*)d_in[m.pbase + 2], (const float*)d_in[m.pbase + 3],
        (const float*)d_in[m.pbase + 5], (const float*)d_in[m.pbase + 6],
        (const float*)d_in[m.pbase + 7], (const float*)d_in[m.pbase + 8],
        (const float*)d_in[m.pbase + 9], (const float*)d_in[m.pbase + 4],
        m.kin, m.k0, m.nout, m.nb3, ws + m.o);
  }
  // 2) init persistent state
  k_fill<<<64, 256, 0, stream>>>((float*)(ws + oU), (size_t)G * 64, 0.f);
  k_fill<<<4, 256, 0, stream>>>((float*)(ws + oCnt), (size_t)G, 0.f);
  k_fill<<<64, 256, 0, stream>>>((float*)(ws + oXg), (size_t)G * 64, 0.f);
  k_count<<<(N + 255) / 256, 256, 0, stream>>>(batch, N, (float*)(ws + oCnt));

  // 3) encoders
  const int nodeTiles = (N + 15) / 16, nodeBlocks = (nodeTiles + WAVES - 1) / WAVES;
  const int edgeTiles = (E + 15) / 16, edgeBlocks = (edgeTiles + WAVES - 1) / WAVES;
  const int globBlocks = ((G / 16) + WAVES - 1) / WAVES;
  k_node_enc<<<nodeBlocks, 128, 0, stream>>>(gx, N, ws + mp[0].o, (float*)(ws + oX));
  k_edge_enc<<<edgeBlocks, 128, 0, stream>>>(gx, ei, ea, N, E, ws + mp[1].o,
                                             (float*)(ws + oE));
  // 4) message-passing layers
  const float INF = __builtin_inff();
  for (int L = 0; L < 3; ++L) {
    const MP& mE = mp[2 + L * 4 + 0];
    const MP& m1 = mp[2 + L * 4 + 1];
    const MP& m2 = mp[2 + L * 4 + 2];
    const MP& mG = mp[2 + L * 4 + 3];
    k_fill<<<2048, 256, 0, stream>>>((float*)(ws + oAgg), (size_t)N * 64, 0.f);
    k_fill<<<64, 256, 0, stream>>>((float*)(ws + oS), (size_t)G * 64, 0.f);
    k_fill<<<64, 256, 0, stream>>>((float*)(ws + oSq), (size_t)G * 64, 0.f);
    k_fill<<<64, 256, 0, stream>>>((float*)(ws + oMi), (size_t)G * 64, INF);
    k_fill<<<64, 256, 0, stream>>>((float*)(ws + oMa), (size_t)G * 64, -INF);
    k_edge_layer<<<edgeBlocks, 128, 0, stream>>>(
        (const float*)(ws + oX), (float*)(ws + oE), (const float*)(ws + oU),
        ei, batch, N, E, ws + mE.o, ws + m1.o, (float*)(ws + oAgg));
    k_node2<<<nodeBlocks, 128, 0, stream>>>(
        (const float*)(ws + oAgg), (const float*)(ws + oU), batch, N, ws + m2.o,
        (float*)(ws + oX), (float*)(ws + oS), (float*)(ws + oSq),
        (float*)(ws + oMi), (float*)(ws + oMa));
    k_glob<<<globBlocks, 128, 0, stream>>>(
        (const float*)(ws + oS), (const float*)(ws + oSq), (const float*)(ws + oMi),
        (const float*)(ws + oMa), (const float*)(ws + oCnt), G, ws + mG.o,
        (float*)(ws + oU));
  }
  // 5) graph readout + decoder
  k_xg<<<8192, 256, 0, stream>>>((const float*)(ws + oX), batch, N,
                                 (float*)(ws + oXg));
  k_decoder<<<globBlocks, 128, 0, stream>>>(
      (const float*)(ws + oXg), (const float*)d_in[NORMG], (const float*)d_in[NORMB],
      G, ws + mp[14].o, (float*)d_out);
}
